// ImprovedHGTEncoder_73126113182035
// MI455X (gfx1250) — compile-verified, matrix operand order checked by reading, not run on hardware
//
#include <hip/hip_runtime.h>
#include <math.h>
#include <stdint.h>

// ---------------------------------------------------------------------------
// HGT encoder for MI455X (gfx1250, wave32, WMMA + TDM).
// GEMMs: each wave computes a 16x64 output strip (4x v_wmma_f32_16x16x32_f16
// per K-step, A fragment reused 4x). B panel (64 cols x Kd) staged into LDS
// by the Tensor Data Mover (tensor_load_to_lds + s_wait_tensorcnt), shared by
// the block's 8 waves. Edge phase: f16 gathers + native float atomics.
// ---------------------------------------------------------------------------

typedef __attribute__((ext_vector_type(16))) _Float16 v16h;
typedef __attribute__((ext_vector_type(8)))  float    v8f;
typedef __attribute__((ext_vector_type(4)))  unsigned v4u;
typedef __attribute__((ext_vector_type(8)))  int      v8i;
typedef __attribute__((ext_vector_type(4)))  int      v4i;

#define NPAPER 100000
#define NAUTHOR 50000
#define NTOT   150000
#define NRELROWS 250000
#define NEDGE  200000
#define NLAYER 3
#define HID_   256
#define DIN_   128
#define H_     4
#define D_     64

// ------------------------------ utility ------------------------------------

__global__ void fill_u32_kernel(unsigned* __restrict__ p, unsigned v, size_t n) {
  size_t i = (size_t)blockIdx.x * blockDim.x + threadIdx.x;
  if (i < n) p[i] = v;
}

// src: [B][K][N] f32 row-major  ->  dst: [B][N][K] f16
__global__ void transpose_f16_kernel(const float* __restrict__ src,
                                     _Float16* __restrict__ dst,
                                     int K, int N, size_t total) {
  size_t idx = (size_t)blockIdx.x * blockDim.x + threadIdx.x;
  if (idx >= total) return;
  size_t kn = (size_t)K * N;
  size_t b  = idx / kn;
  int rem = (int)(idx - b * kn);
  int k = rem / N;
  int n = rem - k * N;
  dst[b * kn + (size_t)n * K + k] = (_Float16)src[idx];
}

__global__ void gelu_kernel(float* __restrict__ x, size_t n) {
  size_t i = (size_t)blockIdx.x * blockDim.x + threadIdx.x;
  if (i >= n) return;
  float v = x[i];
  x[i] = 0.5f * v * (1.0f + erff(v * 0.70710678118654752f));
}

__global__ void copy4_kernel(const float4* __restrict__ s, float4* __restrict__ d, size_t n4) {
  size_t i = (size_t)blockIdx.x * blockDim.x + threadIdx.x;
  if (i < n4) d[i] = s[i];
}

// ------------------------------ WMMA GEMM ----------------------------------
// Y[M,No] = A[M,Kd] @ B[Kd,No] + bias, with Bt pre-transposed f16 [No][Kd].
// Block = 8 waves. Grid: (ceil(mtiles/8), No/64). All waves of a block share
// one 64-col B panel staged in LDS by the TDM; wave w computes M-tile
// (blockIdx.x*8 + w), producing a 16x64 strip with 4 WMMA accumulators.
// Fragment layout per CDNA5 ISA (16-bit, 16x32):
//   half p of lane L holds K = (p<8?0:16) + (L>=16?8:0) + (p&7).
// Requires: mtiles*16 rows valid, Kd % 32 == 0.

template <int KD, typename AT, typename OT>
__global__ void wmma_gemm_tile_kernel(const AT* __restrict__ A, int lda,
                                      const _Float16* __restrict__ Bt,
                                      const float* __restrict__ bias,
                                      OT* __restrict__ Y, int ldo, int mtiles) {
  __shared__ _Float16 Bp[64 * KD];
  const int lane = threadIdx.x & 31;
  const int wave = threadIdx.x >> 5;
  const int n0base = blockIdx.y * 64;

  // ---- stage B panel (64 cols x KD) into LDS via Tensor Data Mover ----
#if __has_builtin(__builtin_amdgcn_tensor_load_to_lds)
  if (threadIdx.x == 0) {
    unsigned lds_off = (unsigned)(uintptr_t)&Bp[0];  // low 32b = LDS byte offset
    unsigned long long ga =
        (unsigned long long)(uintptr_t)(Bt + (size_t)n0base * KD);
    // D# group0: count=1 | lds_addr | global_addr[56:0] | type=2
    v4u g0 = {1u, lds_off, (unsigned)(ga & 0xffffffffu),
              (unsigned)((ga >> 32) & 0x1ffffffu) | (2u << 30)};
    // D# group1: data_size=2B; tensor_dim0=KD; tensor_dim1=64; tile_dim0=KD;
    //            tile_dim1=64; tensor_dim0_stride=KD (rows contiguous)
    v8i g1 = {0x10000,
              (int)((KD & 0xffff) << 16),
              (int)(64 << 16),
              (int)(KD << 16),
              64,
              KD,
              0, 0};
    v4i g2 = {0, 0, 0, 0};
    v4i g3 = {0, 0, 0, 0};
#if defined(__clang_major__) && (__clang_major__ >= 23)
    v8i g4 = {0, 0, 0, 0, 0, 0, 0, 0};
    __builtin_amdgcn_tensor_load_to_lds(g0, g1, g2, g3, g4, 0);
#else
    __builtin_amdgcn_tensor_load_to_lds(g0, g1, g2, g3, 0);
#endif
    __builtin_amdgcn_s_wait_tensorcnt(0);
  }
#else
  for (int i = threadIdx.x; i < 64 * KD; i += blockDim.x)
    Bp[i] = Bt[(size_t)(n0base + i / KD) * KD + (i % KD)];
#endif
  __syncthreads();

  const int mt = blockIdx.x * (int)(blockDim.x >> 5) + wave;
  if (mt >= mtiles) return;

  const int m0 = mt * 16;
  const int r  = lane & 15;
  const int hi = lane >> 4;
  const size_t arow = (size_t)(m0 + r) * (size_t)lda;

  v8f acc0 = {}, acc1 = {}, acc2 = {}, acc3 = {};
  for (int k0 = 0; k0 < KD; k0 += 32) {
    v16h a, b0, b1, b2, b3;
#pragma unroll
    for (int p = 0; p < 16; ++p) {
      int kk = k0 + ((p < 8) ? 0 : 16) + hi * 8 + (p & 7);
      a[p]  = (_Float16)A[arow + kk];
      b0[p] = Bp[(size_t)r * KD + kk];
      b1[p] = Bp[(size_t)(16 + r) * KD + kk];
      b2[p] = Bp[(size_t)(32 + r) * KD + kk];
      b3[p] = Bp[(size_t)(48 + r) * KD + kk];
    }
    acc0 = __builtin_amdgcn_wmma_f32_16x16x32_f16(false, a, false, b0, (short)0, acc0, false, false);
    acc1 = __builtin_amdgcn_wmma_f32_16x16x32_f16(false, a, false, b1, (short)0, acc1, false, false);
    acc2 = __builtin_amdgcn_wmma_f32_16x16x32_f16(false, a, false, b2, (short)0, acc2, false, false);
    acc3 = __builtin_amdgcn_wmma_f32_16x16x32_f16(false, a, false, b3, (short)0, acc3, false, false);
  }

  const float bv0 = bias ? bias[n0base + r]      : 0.0f;
  const float bv1 = bias ? bias[n0base + 16 + r] : 0.0f;
  const float bv2 = bias ? bias[n0base + 32 + r] : 0.0f;
  const float bv3 = bias ? bias[n0base + 48 + r] : 0.0f;
#pragma unroll
  for (int i = 0; i < 8; ++i) {
    const size_t yb = (size_t)(m0 + i + hi * 8) * (size_t)ldo + n0base;
    Y[yb + r]      = (OT)(acc0[i] + bv0);
    Y[yb + 16 + r] = (OT)(acc1[i] + bv1);
    Y[yb + 32 + r] = (OT)(acc2[i] + bv2);
    Y[yb + 48 + r] = (OT)(acc3[i] + bv3);
  }
}

// ------------------------------ edge phase ---------------------------------

__device__ __forceinline__ unsigned f2ord(float f) {
  unsigned u = __float_as_uint(f);
  return (u & 0x80000000u) ? ~u : (u | 0x80000000u);
}
__device__ __forceinline__ float ord2f(unsigned u) {
  unsigned v = (u & 0x80000000u) ? (u & 0x7fffffffu) : ~u;
  return __uint_as_float(v);
}

// alpha[e,h] = scale * p_rel[h] * dot(Q[dst], Krel[src]) over head slice
__global__ void edge_alpha_kernel(const int* __restrict__ ei, int E,
                                  const _Float16* __restrict__ Q, int dstBase,
                                  const _Float16* __restrict__ Krel, int relBase,
                                  const float* __restrict__ prel,
                                  float* __restrict__ alpha) {
  int idx = blockIdx.x * blockDim.x + threadIdx.x;
  if (idx >= E * H_) return;
  int e = idx >> 2, hh = idx & 3;
  int src = ei[e];
  int dst = ei[E + e];
  const _Float16* qp = Q    + (size_t)(dstBase + dst) * HID_ + hh * D_;
  const _Float16* kp = Krel + (size_t)(relBase + src) * HID_ + hh * D_;
  float acc = 0.0f;
#pragma unroll 8
  for (int d = 0; d < D_; ++d) acc += (float)qp[d] * (float)kp[d];
  alpha[idx] = acc * prel[hh] * 0.125f;  // 1/sqrt(64)
}

__global__ void seg_max_kernel(const int* __restrict__ ei, int E,
                               const float* __restrict__ alpha, int dstBase,
                               unsigned* __restrict__ maxb) {
  int idx = blockIdx.x * blockDim.x + threadIdx.x;
  if (idx >= E * H_) return;
  int e = idx >> 2, hh = idx & 3;
  int dst = ei[E + e];
  atomicMax(&maxb[(size_t)(dstBase + dst) * H_ + hh], f2ord(alpha[idx]));
}

// e = exp(alpha - max[dst]); alpha <- e (in place); sum[dst] += e
__global__ void seg_exp_kernel(const int* __restrict__ ei, int E,
                               float* __restrict__ alpha, int dstBase,
                               const unsigned* __restrict__ maxb,
                               float* __restrict__ sumb) {
  int idx = blockIdx.x * blockDim.x + threadIdx.x;
  if (idx >= E * H_) return;
  int e = idx >> 2, hh = idx & 3;
  int dst = ei[E + e];
  size_t s = (size_t)(dstBase + dst) * H_ + hh;
  float ev = expf(alpha[idx] - ord2f(maxb[s]));
  alpha[idx] = ev;
  atomicAdd(&sumb[s], ev);
}

// agg[dst, h*64+d] += w * Vrel[src, h*64+d]; thread = (edge, head, 4-elem chunk)
__global__ void aggregate_kernel(const int* __restrict__ ei, int E,
                                 const float* __restrict__ ebuf,
                                 const float* __restrict__ sumb, int dstBase,
                                 const _Float16* __restrict__ Vrel, int relBase,
                                 float* __restrict__ agg) {
  int idx = blockIdx.x * blockDim.x + threadIdx.x;
  if (idx >= E * H_ * 16) return;
  int c4 = idx & 15;
  int eh = idx >> 4;
  int e = eh >> 2, hh = eh & 3;
  int src = ei[e];
  int dst = ei[E + e];
  size_t drow = (size_t)(dstBase + dst);
  float w = ebuf[eh] / (sumb[drow * H_ + hh] + 1e-16f);
  const _Float16* vp = Vrel + (size_t)(relBase + src) * HID_ + hh * D_ + c4 * 4;
  float* op = agg + drow * HID_ + hh * D_ + c4 * 4;
#pragma unroll
  for (int j = 0; j < 4; ++j) atomicAdd(op + j, (float)vp[j] * w);
}

// -------------------------- fused epilogue ---------------------------------
// h <- relu(LN(g*aout + (2-g)*h)); one wave32 per 256-wide row.
__global__ void combine_ln_kernel(float* __restrict__ h,
                                  const float* __restrict__ aout,
                                  const float* __restrict__ skip2,
                                  const float* __restrict__ lng,
                                  const float* __restrict__ lnb,
                                  int npaper, int ntot) {
  int row = blockIdx.x * (blockDim.x >> 5) + (threadIdx.x >> 5);
  if (row >= ntot) return;
  int lane = threadIdx.x & 31;
  int t = (row < npaper) ? 0 : 1;
  float g = 1.0f / (1.0f + expf(-skip2[t]));
  float z[8];
  float s = 0.0f, s2 = 0.0f;
  size_t base = (size_t)row * HID_;
#pragma unroll
  for (int j = 0; j < 8; ++j) {
    int c = lane + j * 32;
    float zz = g * aout[base + c] + (2.0f - g) * h[base + c];
    z[j] = zz; s += zz; s2 += zz * zz;
  }
#pragma unroll
  for (int off = 16; off > 0; off >>= 1) {
    s  += __shfl_xor(s,  off, 32);
    s2 += __shfl_xor(s2, off, 32);
  }
  float m   = s * (1.0f / 256.0f);
  float var = s2 * (1.0f / 256.0f) - m * m;
  float inv = rsqrtf(var + 1e-5f);
#pragma unroll
  for (int j = 0; j < 8; ++j) {
    int c = lane + j * 32;
    float y = (z[j] - m) * inv * lng[c] + lnb[c];
    h[base + c] = fmaxf(y, 0.0f);
  }
}

// ------------------------------ host side ----------------------------------

extern "C" void kernel_launch(void* const* d_in, const int* in_sizes, int n_in,
                              void* d_out, int out_size, void* d_ws, size_t ws_size,
                              hipStream_t stream) {
  (void)in_sizes; (void)n_in; (void)out_size; (void)ws_size;

  const float* x_paper  = (const float*)d_in[0];
  const float* x_author = (const float*)d_in[1];
  const int*   ei[3]    = { (const int*)d_in[2], (const int*)d_in[3], (const int*)d_in[4] };
  const float* W_in = (const float*)d_in[5];
  const float* b_in = (const float*)d_in[6];
  const float* Wk   = (const float*)d_in[7];
  const float* bk   = (const float*)d_in[8];
  const float* Wq   = (const float*)d_in[9];
  const float* bq   = (const float*)d_in[10];
  const float* Wv   = (const float*)d_in[11];
  const float* bvv  = (const float*)d_in[12];
  const float* Wa   = (const float*)d_in[13];
  const float* ba   = (const float*)d_in[14];
  const float* skip = (const float*)d_in[15];
  const float* a_rel = (const float*)d_in[16];
  const float* m_rel = (const float*)d_in[17];
  const float* p_rel = (const float*)d_in[18];
  const float* ln_g  = (const float*)d_in[19];
  const float* ln_b  = (const float*)d_in[20];

  // relation metadata: RELS = [(1,0),(0,1),(0,0)]
  const int srcRows[3] = {NAUTHOR, NPAPER, NPAPER};
  const int srcBase[3] = {NPAPER, 0, 0};   // row base in node-feature buffers
  const int dstBase[3] = {0, NPAPER, 0};
  const int relOff[3]  = {0, NAUTHOR, NAUTHOR + NPAPER};  // row base in Krel/Vrel

  // ------------------ workspace carve-out ------------------
  char* ws = (char*)d_ws;
  size_t off = 0;
  auto carve = [&](size_t bytes) -> char* {
    char* p = ws + off;
    off = (off + bytes + 255) & ~(size_t)255;
    return p;
  };
  float*    hbuf  = (float*)   carve((size_t)NTOT * HID_ * 4);
  float*    AGG   = (float*)   carve((size_t)NTOT * HID_ * 4);
  float*    AOUT  = (float*)   carve((size_t)NTOT * HID_ * 4);
  _Float16* Kf    = (_Float16*)carve((size_t)NTOT * HID_ * 2);
  _Float16* Qf    = (_Float16*)carve((size_t)NTOT * HID_ * 2);
  _Float16* Vf    = (_Float16*)carve((size_t)NTOT * HID_ * 2);
  _Float16* Krel  = (_Float16*)carve((size_t)NRELROWS * HID_ * 2);
  _Float16* Vrel  = (_Float16*)carve((size_t)NRELROWS * HID_ * 2);
  float*    alphaB= (float*)   carve((size_t)3 * NEDGE * H_ * 4);
  unsigned* maxB  = (unsigned*)carve((size_t)NTOT * H_ * 4);
  float*    sumB  = (float*)   carve((size_t)NTOT * H_ * 4);
  _Float16* WinT  = (_Float16*)carve((size_t)2 * DIN_ * HID_ * 2);
  _Float16* WkT   = (_Float16*)carve((size_t)NLAYER * 2 * HID_ * HID_ * 2);
  _Float16* WqT   = (_Float16*)carve((size_t)NLAYER * 2 * HID_ * HID_ * 2);
  _Float16* WvT   = (_Float16*)carve((size_t)NLAYER * 2 * HID_ * HID_ * 2);
  _Float16* WaT   = (_Float16*)carve((size_t)NLAYER * 2 * HID_ * HID_ * 2);
  _Float16* arelT = (_Float16*)carve((size_t)NLAYER * 3 * H_ * D_ * D_ * 2);
  _Float16* mrelT = (_Float16*)carve((size_t)NLAYER * 3 * H_ * D_ * D_ * 2);

  const dim3 blk256(256);  // 8 waves per block

  // ------------------ weight prep (f16 transpose) ------------------
  auto xT = [&](const float* s, _Float16* d, int B, int K, int N) {
    size_t total = (size_t)B * K * N;
    transpose_f16_kernel<<<(unsigned)((total + 255) / 256), 256, 0, stream>>>(s, d, K, N, total);
  };
  xT(W_in, WinT, 2, DIN_, HID_);
  xT(Wk, WkT, NLAYER * 2, HID_, HID_);
  xT(Wq, WqT, NLAYER * 2, HID_, HID_);
  xT(Wv, WvT, NLAYER * 2, HID_, HID_);
  xT(Wa, WaT, NLAYER * 2, HID_, HID_);
  xT(a_rel, arelT, NLAYER * 3 * H_, D_, D_);
  xT(m_rel, mrelT, NLAYER * 3 * H_, D_, D_);

  const int mtP = NPAPER / 16, mtA = NAUTHOR / 16;
  auto gx = [](int mt) { return (unsigned)((mt + 7) / 8); };

  // ------------------ input projection (K=128) ------------------
  wmma_gemm_tile_kernel<DIN_, float, float><<<dim3(gx(mtP), 4), blk256, 0, stream>>>(
      x_paper, DIN_, WinT, b_in, hbuf, HID_, mtP);
  wmma_gemm_tile_kernel<DIN_, float, float><<<dim3(gx(mtA), 4), blk256, 0, stream>>>(
      x_author, DIN_, WinT + (size_t)HID_ * DIN_, b_in + HID_,
      hbuf + (size_t)NPAPER * HID_, HID_, mtA);

  const int ehThreads = NEDGE * H_;
  const unsigned ehGrid  = (ehThreads + 255) / 256;
  const unsigned aggGrid = ((unsigned)((size_t)ehThreads * 16 + 255) / 256);

  for (int l = 0; l < NLAYER; ++l) {
    // --- K/Q/V projections per node type ---
    for (int t = 0; t < 2; ++t) {
      int mt   = t ? mtA : mtP;
      int ro   = t ? NPAPER : 0;
      const float* Ap = hbuf + (size_t)ro * HID_;
      size_t wOff = (size_t)(l * 2 + t) * HID_ * HID_;
      size_t bOff = (size_t)(l * 2 + t) * HID_;
      wmma_gemm_tile_kernel<HID_, float, _Float16><<<dim3(gx(mt), 4), blk256, 0, stream>>>(
          Ap, HID_, WkT + wOff, bk + bOff, Kf + (size_t)ro * HID_, HID_, mt);
      wmma_gemm_tile_kernel<HID_, float, _Float16><<<dim3(gx(mt), 4), blk256, 0, stream>>>(
          Ap, HID_, WqT + wOff, bq + bOff, Qf + (size_t)ro * HID_, HID_, mt);
      wmma_gemm_tile_kernel<HID_, float, _Float16><<<dim3(gx(mt), 4), blk256, 0, stream>>>(
          Ap, HID_, WvT + wOff, bvv + bOff, Vf + (size_t)ro * HID_, HID_, mt);
    }

    // --- per-relation, per-head 64x64 transforms (K=64) ---
    for (int r = 0; r < 3; ++r) {
      int mt = srcRows[r] / 16;
      for (int hh = 0; hh < H_; ++hh) {
        size_t wOff = (size_t)((l * 3 + r) * H_ + hh) * D_ * D_;
        wmma_gemm_tile_kernel<D_, _Float16, _Float16><<<dim3(gx(mt), 1), blk256, 0, stream>>>(
            Kf + (size_t)srcBase[r] * HID_ + hh * D_, HID_,
            arelT + wOff, (const float*)nullptr,
            Krel + (size_t)relOff[r] * HID_ + hh * D_, HID_, mt);
        wmma_gemm_tile_kernel<D_, _Float16, _Float16><<<dim3(gx(mt), 1), blk256, 0, stream>>>(
            Vf + (size_t)srcBase[r] * HID_ + hh * D_, HID_,
            mrelT + wOff, (const float*)nullptr,
            Vrel + (size_t)relOff[r] * HID_ + hh * D_, HID_, mt);
      }
    }

    // --- edge attention: logits, segment softmax, aggregation ---
    {
      size_t nmx = (size_t)NTOT * H_;
      fill_u32_kernel<<<(unsigned)((nmx + 255) / 256), 256, 0, stream>>>(maxB, 0x007FFFFFu, nmx);  // ord(-inf)
      fill_u32_kernel<<<(unsigned)((nmx + 255) / 256), 256, 0, stream>>>((unsigned*)sumB, 0u, nmx);
      size_t nag = (size_t)NTOT * HID_;
      fill_u32_kernel<<<(unsigned)((nag + 255) / 256), 256, 0, stream>>>((unsigned*)AGG, 0u, nag);
    }
    for (int r = 0; r < 3; ++r)
      edge_alpha_kernel<<<ehGrid, 256, 0, stream>>>(
          ei[r], NEDGE, Qf, dstBase[r], Krel, relOff[r],
          p_rel + (size_t)(l * 3 + r) * H_, alphaB + (size_t)r * ehThreads);
    for (int r = 0; r < 3; ++r)
      seg_max_kernel<<<ehGrid, 256, 0, stream>>>(
          ei[r], NEDGE, alphaB + (size_t)r * ehThreads, dstBase[r], maxB);
    for (int r = 0; r < 3; ++r)
      seg_exp_kernel<<<ehGrid, 256, 0, stream>>>(
          ei[r], NEDGE, alphaB + (size_t)r * ehThreads, dstBase[r], maxB, sumB);
    for (int r = 0; r < 3; ++r)
      aggregate_kernel<<<aggGrid, 256, 0, stream>>>(
          ei[r], NEDGE, alphaB + (size_t)r * ehThreads, sumB, dstBase[r],
          Vrel, relOff[r], AGG);

    // --- output projection: gelu(agg) @ Wa + ba ---
    {
      size_t nag = (size_t)NTOT * HID_;
      gelu_kernel<<<(unsigned)((nag + 255) / 256), 256, 0, stream>>>(AGG, nag);
    }
    for (int t = 0; t < 2; ++t) {
      int mt   = t ? mtA : mtP;
      int ro   = t ? NPAPER : 0;
      size_t wOff = (size_t)(l * 2 + t) * HID_ * HID_;
      size_t bOff = (size_t)(l * 2 + t) * HID_;
      wmma_gemm_tile_kernel<HID_, float, float><<<dim3(gx(mt), 4), blk256, 0, stream>>>(
          AGG + (size_t)ro * HID_, HID_, WaT + wOff, ba + bOff,
          AOUT + (size_t)ro * HID_, HID_, mt);
    }

    // --- skip gate + residual + LayerNorm + ReLU (in place on hbuf) ---
    combine_ln_kernel<<<NTOT / 8, 256, 0, stream>>>(
        hbuf, AOUT, skip + (size_t)l * 2, ln_g + (size_t)l * HID_,
        ln_b + (size_t)l * HID_, NPAPER, NTOT);
  }

  // ------------------ emit output ------------------
  size_t n4 = (size_t)NTOT * HID_ / 4;
  copy4_kernel<<<(unsigned)((n4 + 255) / 256), 256, 0, stream>>>(
      (const float4*)hbuf, (float4*)d_out, n4);
}
